// NetVLAD_61813169324045
// MI455X (gfx1250) — matrix operand order repaired and checked by reading, not compiled, not run on hardware
//
#include <hip/hip_runtime.h>
#include <hip/hip_bf16.h>

// ---------------------------------------------------------------------------
// NetVLAD fused kernel for MI455X (gfx1250, wave32, WMMA bf16 16x16x32 + TDM).
//
// Roofline: 17.2 GFLOP of matmul vs ~280 MB of HBM traffic at 23.3 TB/s
// (~12 us) => memory bound. Strategy: read x exactly once via the Tensor Data
// Mover (async DMA into an LDS stage, overlapped with WMMA compute of the
// previous chunk), keep the normalized bf16 x-tile in LDS for BOTH matmuls,
// accumulate vlad in persistent WMMA C fragments, flush once with fp32
// global atomics.
// ---------------------------------------------------------------------------

typedef __attribute__((ext_vector_type(16))) __bf16         v16bf;
typedef __attribute__((ext_vector_type(16))) unsigned short v16u;
typedef __attribute__((ext_vector_type(8)))  float          v8f;
typedef __attribute__((ext_vector_type(4)))  unsigned int   v4u;
typedef __attribute__((ext_vector_type(8)))  int            v8i;
typedef __attribute__((ext_vector_type(4)))  int            v4i;
typedef unsigned short u16;

#define N_   64
#define D_   256
#define L_   4096
#define K_   64
#define LT   128          // l-columns per chunk
#define CHUNKS 4          // chunks per workgroup -> 512 l per WG, 8 WGs per n
#define XP   136          // padded LDS leading dim for x tile (bank-conflict pad)
#define AP   136          // padded LDS leading dim for attention tile

#if __has_builtin(__builtin_amdgcn_tensor_load_to_lds) && \
    __has_builtin(__builtin_amdgcn_s_wait_tensorcnt)
#define HAVE_TDM 1
#else
#define HAVE_TDM 0
#endif

static __device__ __forceinline__ u16 f2bf(float f) {
  unsigned x = __float_as_uint(f);
  unsigned r = (x + 0x7FFFu + ((x >> 16) & 1u)) >> 16;   // round-to-nearest-even
  return (u16)r;
}
static __device__ __forceinline__ float bf2f(u16 b) {
  return __uint_as_float(((unsigned)b) << 16);
}

#if HAVE_TDM
// Issue a TDM 2D tile load: tile_d0 x tile_d1 elements of 4 bytes, global row
// stride stride0 elements, into LDS at lds_off. D# packed per ISA ch.8:
// group0 = {count, lds_addr, global_addr, type=2}; group1 = {data_size=4B,
// tensor dims, tile dims, dim0 stride}. Remaining groups zero (2D tensor,
// no multicast, no padding, no iteration). 6-arg builtin form (clang-23).
static __device__ __forceinline__ void tdm_load_2d(unsigned lds_off, const void* gptr,
                                                   unsigned tensor_d0, unsigned tensor_d1,
                                                   unsigned tile_d0, unsigned tile_d1,
                                                   unsigned long long stride0) {
  unsigned long long ga = (unsigned long long)(uintptr_t)gptr;
  v4u g0;
  g0[0] = 1u;                                                   // count=1 (valid user D#)
  g0[1] = lds_off;                                              // lds_addr [63:32]
  g0[2] = (unsigned)(ga & 0xFFFFFFFFu);                         // global_addr[31:0]
  g0[3] = (unsigned)((ga >> 32) & 0x01FFFFFFu) | (2u << 30);    // ga[56:32] | type=2
  v8i g1;
  g1[0] = (int)(2u << 16);                                      // data_size=2 (4B), mask=0
  g1[1] = (int)((tensor_d0 & 0xFFFFu) << 16);                   // tensor_dim0[15:0]@48
  g1[2] = (int)(((tensor_d0 >> 16) & 0xFFFFu) |
                ((tensor_d1 & 0xFFFFu) << 16));                 // dim0[31:16], dim1[15:0]
  g1[3] = (int)(((tensor_d1 >> 16) & 0xFFFFu) |
                ((tile_d0 & 0xFFFFu) << 16));                   // dim1[31:16], tile_dim0
  g1[4] = (int)(tile_d1 & 0xFFFFu);                             // tile_dim1, tile_dim2=0
  g1[5] = (int)(unsigned)(stride0 & 0xFFFFFFFFull);             // dim0_stride[31:0]
  g1[6] = (int)(unsigned)((stride0 >> 32) & 0xFFFFull);         // dim0_stride[47:32]
  g1[7] = 0;                                                    // dim1_stride unused (2D)
  v4i z4 = {0, 0, 0, 0};
  v8i z8 = {0, 0, 0, 0, 0, 0, 0, 0};
  __builtin_amdgcn_tensor_load_to_lds(g0, g1, z4, z4, z8, 0);
}
#endif

// A fragment: 16x32 bf16 tile (rows = M, cols = K) from row-major LDS matrix.
// Layout: lane m=lane&15; element e holds K = khalf+e
// (khalf = 16 for lanes 16..31).
static __device__ __forceinline__ v16bf load_a_frag(const u16* __restrict__ base,
                                                    int ld, int row0, int k0, int lane) {
  const int m = lane & 15;
  const int khalf = (lane >> 4) * 16;
  const u16* rowp = base + (row0 + m) * ld + k0 + khalf;
  v16u u;
#pragma unroll
  for (int v = 0; v < 16; ++v) u[v] = rowp[v];
  return __builtin_bit_cast(v16bf, u);
}

// B fragment: 32x16 bf16 tile (rows = K, cols = N) from row-major LDS matrix.
// Lanes 0..15 hold K=0..15, lanes 16..31 hold K=16..31; column n = lane&15.
static __device__ __forceinline__ v16bf load_b_frag(const u16* __restrict__ base,
                                                    int ld, int k0, int col0, int lane) {
  const int n  = lane & 15;
  const int kb = (lane >> 4) * 16;
  v16u u;
#pragma unroll
  for (int e = 0; e < 16; ++e)
    u[e] = base[(k0 + kb + e) * ld + col0 + n];
  return __builtin_bit_cast(v16bf, u);
}

// B fragment of X^T (element [l][d] = Xs[d][l]): contiguous 16 bf16 along l.
static __device__ __forceinline__ v16bf load_bt_frag(const u16* __restrict__ base,
                                                     int ld, int l0, int col0, int lane) {
  const int d  = col0 + (lane & 15);
  const int kb = (lane >> 4) * 16;
  const u16* p = base + d * ld + l0 + kb;
  v16u u;
#pragma unroll
  for (int e = 0; e < 16; ++e) u[e] = p[e];
  return __builtin_bit_cast(v16bf, u);
}

static __device__ __forceinline__ void atomic_add_f32(float* p, float v) {
  __hip_atomic_fetch_add(p, v, __ATOMIC_RELAXED, __HIP_MEMORY_SCOPE_AGENT);
}

// ---------------------------------------------------------------------------
// Kernel 1: per (n, L/8 slice): TDM-prefetch chunk -> normalize -> logits
// (WMMA) -> softmax -> vlad partial (WMMA, persistent accumulators) -> flush.
// ---------------------------------------------------------------------------
__global__ __launch_bounds__(256)
void netvlad_main(const float* __restrict__ x, const float* __restrict__ w,
                  float* __restrict__ vlad_ws, float* __restrict__ asum_ws) {
#if HAVE_TDM
  __shared__ float stage[D_][LT];   // TDM fp32 landing tile     (128 KB)
#endif
  __shared__ u16  Xs[D_][XP];       // normalized x tile, bf16   (~69.6 KB)
  __shared__ u16  Ws[K_][D_];       // conv weights, bf16        (32 KB)
  __shared__ u16  As[K_][AP];       // softmax attentions, bf16  (~17.4 KB)
  __shared__ float colscale[LT];
  __shared__ float red[256];
  __shared__ float asum_s[K_];

  const int tid  = threadIdx.x;
  const int lane = tid & 31;
  const int wv   = tid >> 5;                 // wave id 0..7 (wave32)
  const int g    = blockIdx.x;
  const int n    = g >> 3;
  const int lbase = (g & 7) * (LT * CHUNKS);

  const float* __restrict__ xn = x + (size_t)n * D_ * L_;   // this image

  // stage conv weights as bf16 in LDS (L2-resident, 64 KB fp32 read)
  __builtin_prefetch(w + tid, 0, 1);         // global_prefetch_b8 warm-up
  for (int i = tid; i < K_ * D_; i += 256)
    ((u16*)Ws)[i] = f2bf(w[i]);
  if (tid < K_) asum_s[tid] = 0.f;

#if HAVE_TDM
  // kick off the DMA for chunk 0 (wave 7 owns the TDM pipe; TENSORcnt is
  // per-wave, EXEC is ignored by tensor ops)
  if (wv == 7) {
    tdm_load_2d((unsigned)(uintptr_t)&stage[0][0], xn + lbase,
                /*tensor_d0=*/LT, /*tensor_d1=*/D_,
                /*tile_d0=*/LT, /*tile_d1=*/D_, /*stride0=*/L_);
  }
#endif

  const v8f vzero = {0.f, 0.f, 0.f, 0.f, 0.f, 0.f, 0.f, 0.f};
  v8f vacc[4][2];                            // persistent vlad C fragments
#pragma unroll
  for (int rt = 0; rt < 4; ++rt) { vacc[rt][0] = vzero; vacc[rt][1] = vzero; }

  __syncthreads();                           // Ws ready

  const int j    = tid & (LT - 1);           // column within chunk
  const int half = tid >> 7;                 // which 128 of D this thread scans

  for (int it = 0; it < CHUNKS; ++it) {
    const int l0 = lbase + it * LT;

#if HAVE_TDM
    if (wv == 7) __builtin_amdgcn_s_wait_tensorcnt((short)0);
    __syncthreads();                         // stage[it] visible to all waves
#endif

    // ---- phase 1: per-column sum(x^2) + bf16 convert ----
    float ss = 0.f;
    {
#if HAVE_TDM
#pragma unroll 4
      for (int dd = 0; dd < 128; ++dd) {
        float v = stage[half * 128 + dd][j];
        ss += v * v;
        Xs[half * 128 + dd][j] = f2bf(v);
      }
#else
      const float* __restrict__ xp = xn + (size_t)(half * 128) * L_ + l0 + j;
#pragma unroll 4
      for (int dd = 0; dd < 128; ++dd) {
        float v = xp[(size_t)dd * L_];
        ss += v * v;
        Xs[half * 128 + dd][j] = f2bf(v);
      }
#endif
    }
    red[tid] = ss;
    __syncthreads();                         // all stage reads done
    if (tid < LT) {
      float s = red[tid] + red[tid + LT];
      colscale[tid] = 1.f / fmaxf(sqrtf(s), 1e-12f);  // torch F.normalize eps
    }
#if HAVE_TDM
    // stage buffer is free now: stream next chunk in while we do the WMMAs
    if (wv == 7 && it + 1 < CHUNKS) {
      tdm_load_2d((unsigned)(uintptr_t)&stage[0][0], xn + (l0 + LT),
                  LT, D_, LT, D_, L_);
    }
#endif
    __syncthreads();
    {
      float sc = colscale[j];
      for (int dd = 0; dd < 128; ++dd) {
        u16 b = Xs[half * 128 + dd][j];
        Xs[half * 128 + dd][j] = f2bf(bf2f(b) * sc);
      }
    }
    __syncthreads();

    // ---- phase 2: logits tile, wave wv owns 16 l-columns, all 64 k rows ----
    v8f acc[4];
#pragma unroll
    for (int rt = 0; rt < 4; ++rt) acc[rt] = vzero;
    for (int kk = 0; kk < D_ / 32; ++kk) {
      v16bf b = load_b_frag(&Xs[0][0], XP, kk * 32, wv * 16, lane);
#pragma unroll
      for (int rt = 0; rt < 4; ++rt) {
        v16bf a = load_a_frag(&Ws[0][0], D_, rt * 16, kk * 32, lane);
        acc[rt] = __builtin_amdgcn_wmma_f32_16x16x32_bf16(
            false, a, false, b, (short)0, acc[rt], false, false);
      }
    }

    // ---- softmax over K=64 per column (registers + one half-wave xor) ----
    float mx = -3.4e38f;
#pragma unroll
    for (int rt = 0; rt < 4; ++rt)
#pragma unroll
      for (int r = 0; r < 8; ++r) mx = fmaxf(mx, acc[rt][r]);
    mx = fmaxf(mx, __shfl_xor(mx, 16, 32));
    float se = 0.f;
#pragma unroll
    for (int rt = 0; rt < 4; ++rt)
#pragma unroll
      for (int r = 0; r < 8; ++r) {
        float e = __expf(acc[rt][r] - mx);
        acc[rt][r] = e;
        se += e;
      }
    se += __shfl_xor(se, 16, 32);
    float inv = 1.f / se;
    {
      const int col  = wv * 16 + (lane & 15);
      const int moff = (lane >> 4) * 8;
#pragma unroll
      for (int rt = 0; rt < 4; ++rt)
#pragma unroll
        for (int r = 0; r < 8; ++r)
          As[rt * 16 + moff + r][col] = f2bf(acc[rt][r] * inv);
    }
    __syncthreads();

    // ---- sum_l a[k,l] partials (for the rank-1 centroid correction) ----
    {
      const int k = tid & 63, qu = tid >> 6;
      float s = 0.f;
      for (int jj = 0; jj < 32; ++jj) s += bf2f(As[k][qu * 32 + jj]);
      red[tid] = s;
    }
    __syncthreads();
    if (tid < K_) asum_s[tid] += red[tid] + red[64 + tid] + red[128 + tid] + red[192 + tid];

    // ---- phase 3: vlad += A[64x128] * Xnorm^T[128x256], wave owns 2 d-tiles ----
    for (int kk = 0; kk < LT / 32; ++kk) {
      v16bf b0 = load_bt_frag(&Xs[0][0], XP, kk * 32, (2 * wv) * 16, lane);
      v16bf b1 = load_bt_frag(&Xs[0][0], XP, kk * 32, (2 * wv + 1) * 16, lane);
#pragma unroll
      for (int rt = 0; rt < 4; ++rt) {
        v16bf a = load_a_frag(&As[0][0], AP, rt * 16, kk * 32, lane);
        vacc[rt][0] = __builtin_amdgcn_wmma_f32_16x16x32_bf16(
            false, a, false, b0, (short)0, vacc[rt][0], false, false);
        vacc[rt][1] = __builtin_amdgcn_wmma_f32_16x16x32_bf16(
            false, a, false, b1, (short)0, vacc[rt][1], false, false);
      }
    }
    __syncthreads();   // Xs/As reused next chunk
  }

  // ---- flush persistent accumulators: fp32 atomics straight to L2 ----
  {
    float* vbase   = vlad_ws + (size_t)n * (K_ * D_);
    const int moff = (lane >> 4) * 8;
    const int col  = lane & 15;
#pragma unroll
    for (int rt = 0; rt < 4; ++rt)
#pragma unroll
      for (int dt = 0; dt < 2; ++dt) {
        const int d = (2 * wv + dt) * 16 + col;
#pragma unroll
        for (int r = 0; r < 8; ++r) {
          const int k = rt * 16 + moff + r;
          atomic_add_f32(&vbase[k * D_ + d], vacc[rt][dt][r]);
        }
      }
  }
  if (tid < K_) atomic_add_f32(&asum_ws[n * K_ + tid], asum_s[tid]);
}

// ---------------------------------------------------------------------------
// Kernel 2: vlad = raw - asum*c ; intra-norm over d ; global norm ; write out.
// ---------------------------------------------------------------------------
__global__ __launch_bounds__(256)
void netvlad_finalize(const float* __restrict__ vlad_ws,
                      const float* __restrict__ asum_ws,
                      const float* __restrict__ cent,
                      float* __restrict__ out) {
  __shared__ float buf[K_ * D_];     // 64 KB
  __shared__ float red[256];
  __shared__ float invk_s[K_];
  __shared__ float ksum[K_];
  __shared__ float ginv_s;

  const int n = blockIdx.x, t = threadIdx.x;

  for (int idx = t; idx < K_ * D_; idx += 256) {
    const int k = idx >> 8;
    buf[idx] = vlad_ws[(size_t)n * K_ * D_ + idx] - asum_ws[n * K_ + k] * cent[idx];
  }
  __syncthreads();

  {  // per-cluster sum of squares (4 threads per k)
    const int k = t >> 2, qu = t & 3;
    const float* p = buf + k * D_ + qu * 64;
    float s = 0.f;
    for (int i = 0; i < 64; ++i) { float v = p[i]; s += v * v; }
    red[t] = s;
  }
  __syncthreads();
  if (t < K_) {
    float s  = red[4 * t] + red[4 * t + 1] + red[4 * t + 2] + red[4 * t + 3];
    float iv = 1.f / fmaxf(sqrtf(s), 1e-12f);
    invk_s[t] = iv;
    ksum[t]   = s * iv * iv;          // contribution to global norm
  }
  __syncthreads();
  if (t == 0) {
    float gsum = 0.f;
    for (int k = 0; k < K_; ++k) gsum += ksum[k];
    ginv_s = 1.f / fmaxf(sqrtf(gsum), 1e-12f);
  }
  __syncthreads();

  const float gi = ginv_s;
  for (int idx = t; idx < K_ * D_; idx += 256) {
    const int k = idx >> 8;
    out[(size_t)n * K_ * D_ + idx] = buf[idx] * invk_s[k] * gi;
  }
}

// ---------------------------------------------------------------------------
extern "C" void kernel_launch(void* const* d_in, const int* in_sizes, int n_in,
                              void* d_out, int out_size, void* d_ws, size_t ws_size,
                              hipStream_t stream) {
  const float* x = (const float*)d_in[0];        // [N, D, H, W]
  const float* w = (const float*)d_in[1];        // [K, D]
  const float* c = (const float*)d_in[2];        // [K, D]
  float* out = (float*)d_out;                    // [N, K*D]

  float* vlad_ws = (float*)d_ws;                               // [N, K, D]
  float* asum_ws = vlad_ws + (size_t)N_ * K_ * D_;             // [N, K]
  const size_t zbytes = ((size_t)N_ * K_ * D_ + (size_t)N_ * K_) * sizeof(float);

  (void)hipMemsetAsync(d_ws, 0, zbytes, stream); // accumulators must start at 0
  netvlad_main<<<N_ * 8, 256, 0, stream>>>(x, w, vlad_ws, asum_ws);
  netvlad_finalize<<<N_, 256, 0, stream>>>(vlad_ws, asum_ws, c, out);
}
